// MultiHeadAttention_16338055594447
// MI455X (gfx1250) — matrix operand order repaired and checked by reading, not compiled, not run on hardware
//
#include <hip/hip_runtime.h>

typedef _Float16 half_t;
typedef _Float16 v8h  __attribute__((ext_vector_type(8)));
typedef _Float16 v16h __attribute__((ext_vector_type(16)));
typedef float    v8f  __attribute__((ext_vector_type(8)));

#define NSTATE 1024
#define NHEAD  16
#define HDIM   64
#define BB     2
#define SS     2048
#define QSCALE 0.35355339059327373f /* 64^-0.25 */

// ---------------------------------------------------------------------------
// CDNA5 WMMA f32 = f16 x f16 + f32, 16x16x32.  Fragment mapping used by this code (wave32, contiguous k):
//   A (16x32): lane L holds rows m=L%16, k in [16h, 16h+16)
//              (h = L>>4), packed 2-per-VGPR in order.
//   B (32x16): lane L holds col n=L%16, k in [16h, 16h+16), packed in order.
//   C (16x16): c[v] @ lane L = C[v + 8h][L%16]
// ---------------------------------------------------------------------------
__device__ inline v8f wmma16(v16h a, v16h b, v8f c) {
  return __builtin_amdgcn_wmma_f32_16x16x32_f16(
      /*neg_a=*/false, a, /*neg_b=*/false, b,
      /*c_mod=*/(short)0, c, /*reuse_a=*/false, /*reuse_b=*/false);
}

// A fragment: ptr = row_m_base + kchunk + 16*half (one 32B run)
__device__ inline v16h load_a_frag(const half_t* p) {
  v8h lo = *(const v8h*)(p);
  v8h hi = *(const v8h*)(p + 8);
  return __builtin_shufflevector(lo, hi, 0, 1, 2, 3, 4, 5, 6, 7,
                                 8, 9, 10, 11, 12, 13, 14, 15);
}
// B fragment: ptr = row_n_base + kchunk + 16*half (one 32B run)
__device__ inline v16h load_b_frag(const half_t* p) {
  v8h lo = *(const v8h*)(p);
  v8h hi = *(const v8h*)(p + 8);
  return __builtin_shufflevector(lo, hi, 0, 1, 2, 3, 4, 5, 6, 7,
                                 8, 9, 10, 11, 12, 13, 14, 15);
}

__device__ inline float rowmax16(float x) {
  #pragma unroll
  for (int m = 8; m >= 1; m >>= 1) x = fmaxf(x, __shfl_xor(x, m, 16));
  return x;
}
__device__ inline float rowsum16(float x) {
  #pragma unroll
  for (int m = 8; m >= 1; m >>= 1) x += __shfl_xor(x, m, 16);
  return x;
}

// ---------------------------------------------------------------------------
// Kernel 0: split f32 -> (hi, lo) f16 pair.  x = hi + lo to ~2^-22 relative.
// ---------------------------------------------------------------------------
__global__ void __launch_bounds__(256)
split_kernel(const float* __restrict__ src, half_t* __restrict__ dh,
             half_t* __restrict__ dl, int n) {
  int i = blockIdx.x * 256 + threadIdx.x;
  if (i < n) {
    float x  = src[i];
    half_t h = (half_t)x;
    dh[i]    = h;
    dl[i]    = (half_t)(x - (float)h);
  }
}

// ---------------------------------------------------------------------------
// Kernel 1: QKV projections with split-f16 WMMA.  y = x @ W^T (+b), scale
// folded into q,k.  q/k stored split row-major; v stored split + transposed
// [b][h][d][S] with packed b128 stores.
// grid (M/16, (N/16)/4, 3), block 128.
// ---------------------------------------------------------------------------
__global__ void __launch_bounds__(128)
qkv_proj_kernel(const half_t* __restrict__ xh, const half_t* __restrict__ xl,
                const half_t* __restrict__ Wh_all, const half_t* __restrict__ Wl_all,
                const float* __restrict__ bq, const float* __restrict__ bv,
                half_t* __restrict__ qh, half_t* __restrict__ ql,
                half_t* __restrict__ kh, half_t* __restrict__ kl,
                half_t* __restrict__ vTh, half_t* __restrict__ vTl) {
  const int lane  = threadIdx.x & 31;
  const int wid   = threadIdx.x >> 5;
  const int l15   = lane & 15;
  const int half  = lane >> 4;
  const int mbase = blockIdx.x * 16;
  const int nbase = (blockIdx.y * 4 + wid) * 16;
  const int p     = blockIdx.z;  // 0=q, 1=k, 2=v

  const half_t* Wh = Wh_all + (size_t)p * NSTATE * NSTATE;
  const half_t* Wl = Wl_all + (size_t)p * NSTATE * NSTATE;

  const half_t* arow_h = xh + (size_t)(mbase + l15) * NSTATE + 16 * half;
  const half_t* arow_l = xl + (size_t)(mbase + l15) * NSTATE + 16 * half;
  const half_t* brow_h = Wh + (size_t)(nbase + l15) * NSTATE + 16 * half;
  const half_t* brow_l = Wl + (size_t)(nbase + l15) * NSTATE + 16 * half;

  v8f acc = {};
  #pragma unroll 2
  for (int kc = 0; kc < NSTATE; kc += 32) {
    v16h ah = load_a_frag(arow_h + kc);
    v16h al = load_a_frag(arow_l + kc);
    v16h bh = load_b_frag(brow_h + kc);
    v16h bl = load_b_frag(brow_l + kc);
    acc = wmma16(ah, bh, acc);
    acc = wmma16(ah, bl, acc);
    acc = wmma16(al, bh, acc);
  }

  if (p == 2) {  // v: bias, transpose to [b][h][d][S], packed split store
    const float bval = bv[nbase + l15];
    const int   head = (nbase + l15) >> 6;
    const int   dcol = (nbase + l15) & 63;
    const int   bat  = mbase >> 11;
    const int   s0   = (mbase & 2047) + 8 * half;
    const size_t off = ((size_t)(bat * NHEAD + head) * HDIM + dcol) * SS + s0;
    v8h hp, lp;
    #pragma unroll
    for (int v = 0; v < 8; ++v) {
      float val = acc[v] + bval;
      half_t h  = (half_t)val;
      hp[v]     = h;
      lp[v]     = (half_t)(val - (float)h);
    }
    *(v8h*)(vTh + off) = hp;
    *(v8h*)(vTl + off) = lp;
  } else {       // q or k: (acc + bias) * scale, split row-major
    const float bval = (p == 0) ? bq[nbase + l15] : 0.0f;
    half_t* dh = (p == 0) ? qh : kh;
    half_t* dl = (p == 0) ? ql : kl;
    #pragma unroll
    for (int v = 0; v < 8; ++v) {
      const int m = v + 8 * half;
      float val   = (acc[v] + bval) * QSCALE;
      half_t h    = (half_t)val;
      const size_t off = (size_t)(mbase + m) * NSTATE + nbase + l15;
      dh[off] = h;
      dl[off] = (half_t)(val - (float)h);
    }
  }
}

// ---------------------------------------------------------------------------
// Kernel 2: fused attention.  One wave owns 16 query rows of one (b,h).
// 32-key chunks: 12 score WMMAs (split f16), masked logits streamed to the
// qk output in f32, online softmax + 12 P@V WMMAs (skipped when the chunk is
// entirely above the causal diagonal -- contributions underflow to exactly 0).
// grid (S/16, B*NHEAD), block 32.
// ---------------------------------------------------------------------------
__global__ void __launch_bounds__(32)
attn_kernel(const half_t* __restrict__ qh, const half_t* __restrict__ ql,
            const half_t* __restrict__ kh, const half_t* __restrict__ kl,
            const half_t* __restrict__ vTh, const half_t* __restrict__ vTl,
            const float* __restrict__ mask,
            float* __restrict__ qk_out,
            half_t* __restrict__ wvh, half_t* __restrict__ wvl) {
  __shared__ __align__(16) half_t Ph[16 * 32];
  __shared__ __align__(16) half_t Pl[16 * 32];

  const int lane  = threadIdx.x & 31;
  const int l15   = lane & 15;
  const int half  = lane >> 4;
  const int mbase = blockIdx.x * 16;
  const int bh    = blockIdx.y;
  const int bat   = bh >> 4;
  const int head  = bh & (NHEAD - 1);

  // Loop-invariant Q fragments (K = 64 -> two 32-chunks, hi+lo)
  const half_t* qbh =
      qh + (size_t)(bat * SS + mbase + l15) * NSTATE + head * HDIM + 16 * half;
  const half_t* qbl =
      ql + (size_t)(bat * SS + mbase + l15) * NSTATE + head * HDIM + 16 * half;
  v16h aqh[2], aql[2];
  #pragma unroll
  for (int c = 0; c < 2; ++c) {
    aqh[c] = load_a_frag(qbh + 32 * c);
    aql[c] = load_a_frag(qbl + 32 * c);
  }

  float* qkrow = qk_out + ((size_t)bh * SS + mbase) * SS;
  const half_t* vbh =
      vTh + ((size_t)(bat * NHEAD + head) * HDIM) * SS;  // + dcol*SS + s
  const half_t* vbl =
      vTl + ((size_t)(bat * NHEAD + head) * HDIM) * SS;

  float rM[8], rL[8];
  #pragma unroll
  for (int v = 0; v < 8; ++v) { rM[v] = -3.0e38f; rL[v] = 0.0f; }
  v8f Oacc[4] = {};

  for (int nb = 0; nb < SS; nb += 32) {
    // ---- scores for two 16-col tiles, K = 64, split f16 (3 products) ----
    v8f sc[2];
    #pragma unroll
    for (int c = 0; c < 2; ++c) {
      v8f s = {};
      const half_t* kbh = kh + (size_t)(bat * SS + nb + 16 * c + l15) * NSTATE +
                          head * HDIM + 16 * half;
      const half_t* kbl = kl + (size_t)(bat * SS + nb + 16 * c + l15) * NSTATE +
                          head * HDIM + 16 * half;
      #pragma unroll
      for (int kc = 0; kc < 2; ++kc) {
        v16h bh_f = load_b_frag(kbh + 32 * kc);
        v16h bl_f = load_b_frag(kbl + 32 * kc);
        s = wmma16(aqh[kc], bh_f, s);
        s = wmma16(aqh[kc], bl_f, s);
        s = wmma16(aql[kc], bh_f, s);
      }
      sc[c] = s;
    }

    // ---- mask add + stream f32 logits to qk output (always) ----
    float sm[2][8];
    #pragma unroll
    for (int c = 0; c < 2; ++c)
      #pragma unroll
      for (int v = 0; v < 8; ++v) {
        const int m = v + 8 * half;
        float s = sc[c][v] + mask[(size_t)(mbase + m) * SS + nb + 16 * c + l15];
        sm[c][v] = s;
        qkrow[(size_t)m * SS + nb + 16 * c + l15] = s;
      }

    // ---- softmax + P@V only for chunks touching the causal region ----
    if (nb <= mbase + 15) {
      float alpha[8];
      #pragma unroll
      for (int v = 0; v < 8; ++v) {
        float rmx  = rowmax16(fmaxf(sm[0][v], sm[1][v]));
        float newM = fmaxf(rM[v], rmx);
        alpha[v]   = __expf(rM[v] - newM);
        rM[v]      = newM;
        float p0 = __expf(sm[0][v] - newM);
        float p1 = __expf(sm[1][v] - newM);
        rL[v] = rL[v] * alpha[v] + rowsum16(p0) + rowsum16(p1);
        const int m = v + 8 * half;
        half_t h0 = (half_t)p0;
        half_t h1 = (half_t)p1;
        Ph[m * 32 + l15]      = h0;
        Ph[m * 32 + 16 + l15] = h1;
        Pl[m * 32 + l15]      = (half_t)(p0 - (float)h0);
        Pl[m * 32 + 16 + l15] = (half_t)(p1 - (float)h1);
      }
      __syncthreads();

      #pragma unroll
      for (int d = 0; d < 4; ++d)
        #pragma unroll
        for (int v = 0; v < 8; ++v) Oacc[d][v] *= alpha[v];

      // A = P (16x32) from LDS, split
      v16h aph = load_a_frag(Ph + l15 * 32 + 16 * half);
      v16h apl = load_a_frag(Pl + l15 * 32 + 16 * half);

      // B = V (32 keys x 16 dcols) per d-block from transposed split buffer
      #pragma unroll
      for (int d = 0; d < 4; ++d) {
        const size_t voff = (size_t)(d * 16 + l15) * SS + nb + 16 * half;
        v16h vh_f = load_b_frag(vbh + voff);
        v16h vl_f = load_b_frag(vbl + voff);
        Oacc[d] = wmma16(aph, vh_f, Oacc[d]);
        Oacc[d] = wmma16(aph, vl_f, Oacc[d]);
        Oacc[d] = wmma16(apl, vh_f, Oacc[d]);
      }
      __syncthreads();
    }
  }

  // ---- finalize: divide by row sums, write wv split row-major ----
  #pragma unroll
  for (int d = 0; d < 4; ++d)
    #pragma unroll
    for (int v = 0; v < 8; ++v) {
      const int m = v + 8 * half;
      float o  = Oacc[d][v] / rL[v];
      half_t h = (half_t)o;
      const size_t off =
          (size_t)(bat * SS + mbase + m) * NSTATE + head * HDIM + d * 16 + l15;
      wvh[off] = h;
      wvl[off] = (half_t)(o - (float)h);
    }
}

// ---------------------------------------------------------------------------
// Kernel 3: out = wv @ Wo^T + bo, split-f16 WMMA, f32 output.
// grid (M/16, (N/16)/4), block 128.
// ---------------------------------------------------------------------------
__global__ void __launch_bounds__(128)
out_proj_kernel(const half_t* __restrict__ wvh, const half_t* __restrict__ wvl,
                const half_t* __restrict__ Woh, const half_t* __restrict__ Wol,
                const float* __restrict__ bo, float* __restrict__ out) {
  const int lane  = threadIdx.x & 31;
  const int wid   = threadIdx.x >> 5;
  const int l15   = lane & 15;
  const int half  = lane >> 4;
  const int mbase = blockIdx.x * 16;
  const int nbase = (blockIdx.y * 4 + wid) * 16;

  const half_t* arow_h = wvh + (size_t)(mbase + l15) * NSTATE + 16 * half;
  const half_t* arow_l = wvl + (size_t)(mbase + l15) * NSTATE + 16 * half;
  const half_t* brow_h = Woh + (size_t)(nbase + l15) * NSTATE + 16 * half;
  const half_t* brow_l = Wol + (size_t)(nbase + l15) * NSTATE + 16 * half;

  v8f acc = {};
  #pragma unroll 2
  for (int kc = 0; kc < NSTATE; kc += 32) {
    v16h ah = load_a_frag(arow_h + kc);
    v16h al = load_a_frag(arow_l + kc);
    v16h bh = load_b_frag(brow_h + kc);
    v16h bl = load_b_frag(brow_l + kc);
    acc = wmma16(ah, bh, acc);
    acc = wmma16(ah, bl, acc);
    acc = wmma16(al, bh, acc);
  }

  const float bval = bo[nbase + l15];
  #pragma unroll
  for (int v = 0; v < 8; ++v) {
    const int m = v + 8 * half;
    out[(size_t)(mbase + m) * NSTATE + nbase + l15] = acc[v] + bval;
  }
}

// ---------------------------------------------------------------------------
extern "C" void kernel_launch(void* const* d_in, const int* in_sizes, int n_in,
                              void* d_out, int out_size, void* d_ws, size_t ws_size,
                              hipStream_t stream) {
  const float* x    = (const float*)d_in[0];
  const float* mask = (const float*)d_in[1];
  const float* Wq   = (const float*)d_in[2];
  const float* bq   = (const float*)d_in[3];
  const float* Wk   = (const float*)d_in[4];
  const float* Wv   = (const float*)d_in[5];
  const float* bv   = (const float*)d_in[6];
  const float* Wo   = (const float*)d_in[7];
  const float* bo   = (const float*)d_in[8];

  const size_t nX = (size_t)BB * SS * NSTATE;  // 4,194,304
  const size_t nW = (size_t)NSTATE * NSTATE;   // 1,048,576

  float* out = (float*)d_out;
  float* qk  = out + nX;  // [B,H,S,S]

  // workspace layout (element offsets in half_t)
  half_t* w   = (half_t*)d_ws;
  half_t* xh  = w;                 // 4M
  half_t* xl  = xh + nX;           // 4M
  half_t* Wh  = xl + nX;           // 3*1M (q,k,v hi, contiguous by p)
  half_t* Wl  = Wh + 3 * nW;       // 3*1M
  half_t* Woh = Wl + 3 * nW;       // 1M
  half_t* Wol = Woh + nW;          // 1M
  half_t* qh  = Wol + nW;          // 4M
  half_t* ql  = qh + nX;
  half_t* kh  = ql + nX;
  half_t* kl  = kh + nX;
  half_t* vTh = kl + nX;           // transposed [b][h][d][S]
  half_t* vTl = vTh + nX;
  half_t* wvh = vTl + nX;
  half_t* wvl = wvh + nX;          // end: 48M half = 96 MB

  // 0) split f32 inputs into f16 hi/lo pairs
  split_kernel<<<(int)(nX / 256), 256, 0, stream>>>(x, xh, xl, (int)nX);
  split_kernel<<<(int)(nW / 256), 256, 0, stream>>>(Wq, Wh + 0 * nW, Wl + 0 * nW, (int)nW);
  split_kernel<<<(int)(nW / 256), 256, 0, stream>>>(Wk, Wh + 1 * nW, Wl + 1 * nW, (int)nW);
  split_kernel<<<(int)(nW / 256), 256, 0, stream>>>(Wv, Wh + 2 * nW, Wl + 2 * nW, (int)nW);
  split_kernel<<<(int)(nW / 256), 256, 0, stream>>>(Wo, Woh, Wol, (int)nW);

  // 1) q/k/v projections (scale folded into q,k; v transposed)
  qkv_proj_kernel<<<dim3((BB * SS) / 16, (NSTATE / 16) / 4, 3), 128, 0, stream>>>(
      xh, xl, Wh, Wl, bq, bv, qh, ql, kh, kl, vTh, vTl);

  // 2) fused attention: f32 logits -> qk output, flash softmax, P@V -> wv
  attn_kernel<<<dim3(SS / 16, BB * NHEAD), 32, 0, stream>>>(
      qh, ql, kh, kl, vTh, vTl, mask, qk, wvh, wvl);

  // 3) output projection
  out_proj_kernel<<<dim3((BB * SS) / 16, (NSTATE / 16) / 4), 128, 0, stream>>>(
      wvh, wvl, Woh, Wol, bo, out);
}
